// NeRFtinymodel_53721450938414
// MI455X (gfx1250) — matrix operand order repaired and checked by reading, not compiled, not run on hardware
//
#include <hip/hip_runtime.h>
#include <hip/hip_bf16.h>

typedef _Float16 f16;
typedef __attribute__((ext_vector_type(16))) _Float16 v16h;
typedef __attribute__((ext_vector_type(8)))  _Float16 v8h;
typedef __attribute__((ext_vector_type(4)))  _Float16 v4h;
typedef __attribute__((ext_vector_type(8)))  float    v8f;
typedef __attribute__((ext_vector_type(4)))  float    v4f;

#define WAVES_PER_BLOCK 8
#define THREADS (WAVES_PER_BLOCK * 32)

// Activation LDS tile: 16 rows x 72 f16 (stride 72 = 36 dwords -> conflict-free,
// 144B row pitch keeps every b128 access 16B-aligned).
constexpr int AS  = 72;
// Weight LDS K-stride for K<=64 layers (same conflict-free/alignment property).
constexpr int WS  = 72;
// W_in (K=32) K-stride.
constexpr int WSI = 40;

// A fragment (16x32 f16, MxK): lane = row (l15), half selects K subsets
// {half*8..half*8+7} and {16+half*8..16+half*8+7}: two contiguous 16B runs.
__device__ __forceinline__ v16h ldsA(const f16* act, int l15, int hf, int kBase) {
  const f16* p = act + l15 * AS + kBase + hf * 8;
  union { v16h v; v8h h[2]; } u;
  u.h[0] = *(const v8h*)(p);
  u.h[1] = *(const v8h*)(p + 16);
  return u.v;
}

// B fragment (32x16 f16, KxN) from transposed weights wT[n][k]:
// lane = column n, K subsets {half*8..half*8+7} and {16+half*8..16+half*8+7}.
__device__ __forceinline__ v16h ldsB(const f16* wT, int n, int hf, int kBase, int stride) {
  const f16* p = wT + n * stride + kBase + hf * 8;
  union { v16h v; v8h h[2]; } u;
  u.h[0] = *(const v8h*)(p);
  u.h[1] = *(const v8h*)(p + 16);
  return u.v;
}

__device__ __forceinline__ v8f wmma_f16(v16h a, v16h b, v8f c) {
  return __builtin_amdgcn_wmma_f32_16x16x32_f16(false, a, false, b, (short)0, c, false, false);
}

// D fragment (16x16 f32): lane = column N (l15), element g -> row M = g + 8*half.
// Fuse bias add + ReLU + f32->f16 into the writeback (keeps WMMA C at literal 0).
__device__ __forceinline__ void storeActFragBR(f16* act, v8f d, float bias,
                                               int l15, int hf, int colBase) {
  int m0 = hf * 8;
#pragma unroll
  for (int g = 0; g < 8; ++g) {
    float v = __builtin_fmaxf(d[g] + bias, 0.f);   // single v_max_num_f32
    act[(m0 + g) * AS + colBase + l15] = (f16)v;
  }
}

// Dense layer, 64 outputs, K = KCH*32 (A already staged in act cols 0..K-1).
template<int KCH>
__device__ __forceinline__ void layer64(f16* act, const f16* wT, int wstride,
                                        const float* bias4, int l15, int hf) {
  v16h a[KCH];
#pragma unroll
  for (int c = 0; c < KCH; ++c) a[c] = ldsA(act, l15, hf, c * 32);
  v8f acc[4];
#pragma unroll
  for (int f = 0; f < 4; ++f) {
    v8f zc = {};                                   // lowers to inline 0 C operand
    acc[f] = wmma_f16(a[0], ldsB(wT, f * 16 + l15, hf, 0, wstride), zc);
    if (KCH == 2)
      acc[f] = wmma_f16(a[1], ldsB(wT, f * 16 + l15, hf, 32, wstride), acc[f]);
  }
  // All LDS reads issued before the writeback (per-wave DS ordering).
#pragma unroll
  for (int f = 0; f < 4; ++f)
    storeActFragBR(act, acc[f], bias4[f], l15, hf, f * 16);
}

// Dense layer, 16 outputs, K = 64, no ReLU; returns D fragment (bias applied).
__device__ __forceinline__ v8f layer16(f16* act, const f16* wT, float bias, int l15, int hf) {
  v16h a0 = ldsA(act, l15, hf, 0);
  v16h a1 = ldsA(act, l15, hf, 32);
  v8f zc = {};
  v8f acc = wmma_f16(a0, ldsB(wT, l15, hf, 0,  WS), zc);
  acc = wmma_f16(a1, ldsB(wT, l15, hf, 32, WS), acc);
#pragma unroll
  for (int g = 0; g < 8; ++g) acc[g] += bias;
  return acc;
}

__global__ __launch_bounds__(THREADS)
void nerf_fused_wmma(const float* __restrict__ emb, const float* __restrict__ enc,
                     const float* __restrict__ Win, const float* __restrict__ bin,
                     const float* __restrict__ W0,  const float* __restrict__ b0,
                     const float* __restrict__ Wd,  const float* __restrict__ bd,
                     const float* __restrict__ Wc,  const float* __restrict__ bc,
                     const float* __restrict__ W1a, const float* __restrict__ b1a,
                     const float* __restrict__ W1b, const float* __restrict__ b1b,
                     const float* __restrict__ Wo,  const float* __restrict__ bo,
                     float* __restrict__ out, int npts) {
  __shared__ __align__(16) f16 sWin[64 * WSI];
  __shared__ __align__(16) f16 sW0 [64 * WS];
  __shared__ __align__(16) f16 sWd [16 * WS];
  __shared__ __align__(16) f16 sWc [64 * WS];
  __shared__ __align__(16) f16 sW1a[64 * WS];
  __shared__ __align__(16) f16 sW1b[64 * WS];
  __shared__ __align__(16) f16 sWo [16 * WS];
  __shared__ __align__(16) f16 sAct[WAVES_PER_BLOCK * 16 * AS];

  const int tid = threadIdx.x;

  // ---- Per-block weight preload: zero (covers K-padding), then fill transposed f16.
  auto zero = [&](f16* p, int n) {
    for (int i = tid; i < n; i += THREADS) p[i] = (f16)0;
  };
  zero(sWin, 64 * WSI); zero(sW0, 64 * WS); zero(sWd, 16 * WS); zero(sWc, 64 * WS);
  zero(sW1a, 64 * WS);  zero(sW1b, 64 * WS); zero(sWo, 16 * WS);
  __syncthreads();
  for (int i = tid; i < 32 * 64; i += THREADS) sWin[(i & 63) * WSI + (i >> 6)] = (f16)Win[i];
  for (int i = tid; i < 64 * 64; i += THREADS) sW0 [(i & 63) * WS  + (i >> 6)] = (f16)W0[i];
  for (int i = tid; i < 64 * 16; i += THREADS) sWd [(i & 15) * WS  + (i >> 4)] = (f16)Wd[i];
  for (int i = tid; i < 54 * 64; i += THREADS) sWc [(i & 63) * WS  + (i >> 6)] = (f16)Wc[i];
  for (int i = tid; i < 64 * 64; i += THREADS) sW1a[(i & 63) * WS  + (i >> 6)] = (f16)W1a[i];
  for (int i = tid; i < 64 * 64; i += THREADS) sW1b[(i & 63) * WS  + (i >> 6)] = (f16)W1b[i];
  for (int i = tid; i < 64 * 3;  i += THREADS) sWo [(i % 3)  * WS  + (i / 3)]  = (f16)Wo[i];
  __syncthreads();

  const int wave = tid >> 5, lane = tid & 31;
  const int l15 = lane & 15, hf = lane >> 4;
  f16* act = sAct + wave * 16 * AS;

  // ---- Per-lane bias preload (lane owns output column N = f*16 + l15).
  float Bin[4], B0[4], Bc[4], B1a[4], B1b[4];
#pragma unroll
  for (int f = 0; f < 4; ++f) {
    Bin[f] = bin[f * 16 + l15];
    B0 [f] = b0 [f * 16 + l15];
    Bc [f] = bc [f * 16 + l15];
    B1a[f] = b1a[f * 16 + l15];
    B1b[f] = b1b[f * 16 + l15];
  }
  const float Bd = bd[l15];
  const float Bo = (l15 < 3) ? bo[l15] : 0.f;

  const int tiles = npts >> 4;
  const int gw = blockIdx.x * WAVES_PER_BLOCK + wave;
  const int nw = gridDim.x * WAVES_PER_BLOCK;

  for (int t = gw; t < tiles; t += nw) {
    const int base = t << 4;

    // Prefetch next tile's points (global_prefetch_b8, speculative).
    const int tn = t + nw;
    if (tn < tiles) {
      __builtin_prefetch(emb + (size_t)tn * 16 * 32 + (size_t)lane * 16, 0, 1);
      __builtin_prefetch(enc + (size_t)tn * 16 * 39 + (size_t)lane * 16, 0, 1);
    }

    // ---- Stage emb tile: 16x32 f32 (contiguous 2KB, streamed NT) -> f16 cols 0..31.
    const float* ebase = emb + (size_t)base * 32;
#pragma unroll
    for (int i = 0; i < 4; ++i) {
      int e = i * 128 + lane * 4;          // flat element in 16x32 tile
      v4f tv = __builtin_nontemporal_load((const v4f*)(ebase + e));
      v4h hv;
#pragma unroll
      for (int j = 0; j < 4; ++j) hv[j] = (f16)tv[j];
      *(v4h*)(act + (e >> 5) * AS + (e & 31)) = hv;
    }

    // ---- MLP trunk.
    layer64<1>(act, sWin, WSI, Bin, l15, hf);          // relu(emb @ W_in + b_in)
    layer64<2>(act, sW0,  WS,  B0,  l15, hf);          // relu(x @ W0 + b0)
    v8f d3 = layer16(act, sWd, Bd, l15, hf);           // x @ Wd + bd (col0=dense, 1..15=h)

    // ---- Stage layer-4 input: [h(15) | enc_dir(39) | zeros(10)].
    if (l15 >= 1) {
#pragma unroll
      for (int g = 0; g < 8; ++g)
        act[(hf * 8 + g) * AS + (l15 - 1)] = (f16)d3[g];
    }
    // enc tile: 16x39 f32 = contiguous, 16B-aligned 2496B run; stream as b128 NT.
    {
      const float* encBase = enc + (size_t)base * 39;
#pragma unroll
      for (int i = 0; i < 5; ++i) {
        int e = i * 128 + lane * 4;        // flat element in 16x39 tile
        if (e < 16 * 39) {
          v4f tv = __builtin_nontemporal_load((const v4f*)(encBase + e));
#pragma unroll
          for (int j = 0; j < 4; ++j) {
            int idx = e + j;
            int r = idx / 39, c = idx - r * 39;
            act[r * AS + 15 + c] = (f16)tv[j];
          }
        }
      }
    }
    if (lane < 16) {
      v8h z = {};
      *(v8h*)(act + lane * AS + 56) = z;   // cols 56..63
      act[lane * AS + 54] = (f16)0;
      act[lane * AS + 55] = (f16)0;
    }

    layer64<2>(act, sWc,  WS, Bc,  l15, hf);           // relu(concat @ Wc + bc)
    layer64<2>(act, sW1a, WS, B1a, l15, hf);           // relu(h @ W1a + b1a)
    layer64<2>(act, sW1b, WS, B1b, l15, hf);           // relu(h @ W1b + b1b)
    v8f d7 = layer16(act, sWo, Bo, l15, hf);           // h @ Wo + bo (cols 0..2)

    // ---- Output: stage [color(3) | dense(1)] f32 in LDS, coalesced NT 16B stores.
    float* os = (float*)act;
    if (l15 < 3) {
#pragma unroll
      for (int g = 0; g < 8; ++g) os[(hf * 8 + g) * 4 + l15] = d7[g];
    }
    if (l15 == 0) {
#pragma unroll
      for (int g = 0; g < 8; ++g) os[(hf * 8 + g) * 4 + 3] = d3[g];
    }
    if (lane < 16) {
      v4f o = *(const v4f*)(os + lane * 4);
      __builtin_nontemporal_store(o, (v4f*)(out + (size_t)(base + lane) * 4));
    }
  }
}

extern "C" void kernel_launch(void* const* d_in, const int* in_sizes, int n_in,
                              void* d_out, int out_size, void* d_ws, size_t ws_size,
                              hipStream_t stream) {
  (void)n_in; (void)out_size; (void)d_ws; (void)ws_size;
  const float* emb = (const float*)d_in[0];
  const float* enc = (const float*)d_in[1];
  const float* Win = (const float*)d_in[2];
  const float* bin = (const float*)d_in[3];
  const float* W0  = (const float*)d_in[4];
  const float* b0  = (const float*)d_in[5];
  const float* Wd  = (const float*)d_in[6];
  const float* bd  = (const float*)d_in[7];
  const float* Wc  = (const float*)d_in[8];
  const float* bc  = (const float*)d_in[9];
  const float* W1a = (const float*)d_in[10];
  const float* b1a = (const float*)d_in[11];
  const float* W1b = (const float*)d_in[12];
  const float* b1b = (const float*)d_in[13];
  const float* Wo  = (const float*)d_in[14];
  const float* bo  = (const float*)d_in[15];
  float* out = (float*)d_out;

  const int npts  = in_sizes[0] / 32;       // emb_points is [N,32]
  const int tiles = npts / 16;
  int blocks = (tiles + WAVES_PER_BLOCK - 1) / WAVES_PER_BLOCK;
  if (blocks > 1024) blocks = 1024;         // persistent grid; amortize weight preload
  if (blocks < 1) blocks = 1;

  nerf_fused_wmma<<<blocks, THREADS, 0, stream>>>(
      emb, enc, Win, bin, W0, b0, Wd, bd, Wc, bc,
      W1a, b1a, W1b, b1b, Wo, bo, out, npts);
}
